// Spatio_Temporal_Att_18004502905264
// MI455X (gfx1250) — matrix operand order repaired and checked by reading, not compiled, not run on hardware
//
#include <hip/hip_runtime.h>

typedef __attribute__((ext_vector_type(16))) _Float16 v16h;
typedef __attribute__((ext_vector_type(8)))  _Float16 v8h;
typedef __attribute__((ext_vector_type(8)))  float    v8f;

constexpr int kB = 16, kT = 64, kN = 256, kC = 16, kHID = 128, kNSQ = 8;
constexpr float kScl = 0.08838834764831845f; // 1/sqrt(128)

__device__ __forceinline__ v8f wmma_f16(v16h a, v16h b, v8f c) {
  return __builtin_amdgcn_wmma_f32_16x16x32_f16(false, a, false, b, (short)0, c,
                                                false, false);
}

// A fragment: 16x32 f16, row-major source, 16B-aligned vector loads.
// lanes 0-15: M=lane,   K = [0..7] in v0-3, [16..23] in v4-7
// lanes16-31: M=lane-16,K = [8..15] in v0-3, [24..31] in v4-7
__device__ __forceinline__ v16h load_afrag(const _Float16* src, int ld, int m0,
                                           int k0) {
  int lane = threadIdx.x & 31;
  const _Float16* p = src + (m0 + (lane & 15)) * ld + k0 + ((lane >> 4) << 3);
  v8h lo = *(const v8h*)p;
  v8h hi = *(const v8h*)(p + 16);
  return __builtin_shufflevector(lo, hi, 0, 1, 2, 3, 4, 5, 6, 7, 8, 9, 10, 11,
                                 12, 13, 14, 15);
}

// B fragment 32x16 (KxN) from K-contiguous source: B[kk][n] = src[n*ld + kk].
// lane gives N; K = [0..7],[16..23] (lanes 0-15) / [8..15],[24..31] (lanes 16-31)
__device__ __forceinline__ v16h load_bfragT(const _Float16* src, int ld, int k0,
                                            int n0) {
  int lane = threadIdx.x & 31;
  const _Float16* p = src + (n0 + (lane & 15)) * ld + k0 + ((lane >> 4) << 3);
  v8h lo = *(const v8h*)p;
  v8h hi = *(const v8h*)(p + 16);
  return __builtin_shufflevector(lo, hi, 0, 1, 2, 3, 4, 5, 6, 7, 8, 9, 10, 11,
                                 12, 13, 14, 15);
}

// ---------------- att0: spatial attention over N per (b,t) ----------------
__global__ __launch_bounds__(256) void att0_kernel(
    const float* __restrict__ x, const float* __restrict__ Wq,
    const float* __restrict__ bq, const float* __restrict__ Wk,
    const float* __restrict__ bk, const float* __restrict__ Wv,
    const float* __restrict__ bv, const float* __restrict__ Ws,
    const float* __restrict__ bs, const float* __restrict__ g_,
    const float* __restrict__ b_, _Float16* __restrict__ hout) {
  __shared__ __attribute__((aligned(16))) _Float16 xs[kN * 32];       // xp, K pad 20->32
  __shared__ __attribute__((aligned(16))) _Float16 wlsT[4][kHID * 32]; // W^T: [c][k]
  __shared__ float bls[4][kHID];
  __shared__ __attribute__((aligned(16))) _Float16 qs[kN * kHID];     // row-major
  __shared__ __attribute__((aligned(16))) _Float16 ks[kN * kHID];     // row-major
  __shared__ __attribute__((aligned(16))) _Float16 vsT[kHID * kN];    // v^T: [c][key]
  __shared__ float mcol[kN];
  __shared__ float zcol[kN];
  __shared__ float lng[kHID];
  __shared__ float lnb[kHID];
  __shared__ __attribute__((aligned(16))) _Float16 pst[8][16 * 32];   // per-wave

  const int tid = threadIdx.x;
  const int wave = tid >> 5, lane = tid & 31;
  const int cN = lane & 15;
  const int rOff = (lane >> 4) << 3;
  const int bt = blockIdx.x;
  const int b = bt / kT, t = bt % kT;

  {  // one xp row per thread: x concat positional encoding
    float tf = (float)t;
    const float* xr = x + ((size_t)bt * kN + tid) * kC;
    _Float16* d = xs + tid * 32;
#pragma unroll
    for (int c = 0; c < kC; ++c) d[c] = (_Float16)xr[c];
    d[16] = (_Float16)__sinf(tf);
    d[17] = (_Float16)__cosf(tf);
    d[18] = (_Float16)__sinf(tf * 0.01f);
    d[19] = (_Float16)__cosf(tf * 0.01f);
#pragma unroll
    for (int c = 20; c < 32; ++c) d[c] = (_Float16)0.0f;
  }
  {
    const float* Wp[4] = {Wq, Wk, Wv, Ws};
    const float* bp[4] = {bq, bk, bv, bs};
#pragma unroll
    for (int m = 0; m < 4; ++m) {
      for (int i = tid; i < 32 * kHID; i += 256) {  // i = r*128 + c, coalesced read
        int r = i >> 7, c = i & 127;
        wlsT[m][c * 32 + r] = (r < kC + 4) ? (_Float16)Wp[m][i] : (_Float16)0.0f;
      }
      if (tid < kHID) bls[m][tid] = bp[m][tid];
    }
    if (tid < kHID) { lng[tid] = g_[tid]; lnb[tid] = b_[tid]; }
  }
  __syncthreads();

  // Q/K/V GEMMs: 3 mats * 16 mtiles * 8 ctiles = 384 WMMA tiles, K=32 one-shot
  for (int tI = wave; tI < 384; tI += 8) {
    int m = tI >> 7, tt = tI & 127;
    int mt = tt >> 3, ct = tt & 7;
    v16h A = load_afrag(xs, 32, mt * 16, 0);
    v16h Bf = load_bfragT(wlsT[m], 32, 0, ct * 16);
    float bias = bls[m][ct * 16 + cN];
    v8f acc;
#pragma unroll
    for (int r = 0; r < 8; ++r) acc[r] = bias;
    acc = wmma_f16(A, Bf, acc);
    if (m < 2) {
      _Float16* dst = (m == 0) ? qs : ks;
#pragma unroll
      for (int r = 0; r < 8; ++r)
        dst[(mt * 16 + r + rOff) * kHID + ct * 16 + cN] = (_Float16)acc[r];
    } else {  // v transposed: [c][key], keys contiguous -> one b128 store
      v8h pk;
#pragma unroll
      for (int r = 0; r < 8; ++r) pk[r] = (_Float16)acc[r];
      *(v8h*)(vsT + (ct * 16 + cN) * kN + mt * 16 + rOff) = pk;
    }
  }
  __syncthreads();

  // per-KEY softmax stats over all 256 queries (softmax axis=-2), online
  for (int kt = wave; kt < 16; kt += 8) {
    float mloc = -3.0e38f, zloc = 0.0f;
    for (int qt = 0; qt < 16; ++qt) {
      v8f s;
#pragma unroll
      for (int r = 0; r < 8; ++r) s[r] = 0.0f;
#pragma unroll
      for (int cc = 0; cc < 4; ++cc) {
        v16h A = load_afrag(qs, kHID, qt * 16, cc * 32);
        v16h Bf = load_bfragT(ks, kHID, cc * 32, kt * 16);
        s = wmma_f16(A, Bf, s);
      }
      float sm = -3.0e38f;
#pragma unroll
      for (int r = 0; r < 8; ++r) { s[r] *= kScl; sm = fmaxf(sm, s[r]); }
      float mn = fmaxf(mloc, sm);
      float zz = zloc * __expf(mloc - mn);
#pragma unroll
      for (int r = 0; r < 8; ++r) zz += __expf(s[r] - mn);
      mloc = mn; zloc = zz;
    }
    float mo = __shfl_xor(mloc, 16, 32);
    float zo = __shfl_xor(zloc, 16, 32);
    float M = fmaxf(mloc, mo);
    float Z = zloc * __expf(mloc - M) + zo * __expf(mo - M);
    if (lane < 16) { mcol[kt * 16 + lane] = M; zcol[kt * 16 + lane] = Z; }
  }
  __syncthreads();

  // pre-scale v by 1/Z(key): vsT[c][k] /= Z[k]
  for (int i = tid; i < kHID * kN; i += 256)
    vsT[i] = (_Float16)((float)vsT[i] / zcol[i & (kN - 1)]);
  __syncthreads();

  // out = p @ v' + xp @ Wskip + bias; then LayerNorm + ELU; store transposed
  for (int qt = wave; qt < 16; qt += 8) {
    v8f acc[8];
#pragma unroll
    for (int ct = 0; ct < 8; ++ct)
#pragma unroll
      for (int r = 0; r < 8; ++r) acc[ct][r] = 0.0f;
    for (int kc = 0; kc < 8; ++kc) {
#pragma unroll
      for (int h = 0; h < 2; ++h) {
        int kt = kc * 2 + h;
        v8f s;
#pragma unroll
        for (int r = 0; r < 8; ++r) s[r] = 0.0f;
#pragma unroll
        for (int cc = 0; cc < 4; ++cc) {
          v16h A = load_afrag(qs, kHID, qt * 16, cc * 32);
          v16h Bf = load_bfragT(ks, kHID, cc * 32, kt * 16);
          s = wmma_f16(A, Bf, s);
        }
        float mc = mcol[kt * 16 + cN];
#pragma unroll
        for (int r = 0; r < 8; ++r)
          pst[wave][(r + rOff) * 32 + h * 16 + cN] =
              (_Float16)__expf(s[r] * kScl - mc);
      }
      v16h pA = load_afrag(pst[wave], 32, 0, 0);
#pragma unroll
      for (int ct = 0; ct < 8; ++ct) {
        v16h Bf = load_bfragT(vsT, kN, kc * 32, ct * 16);
        acc[ct] = wmma_f16(pA, Bf, acc[ct]);
      }
    }
    {  // skip connection
      v16h A = load_afrag(xs, 32, qt * 16, 0);
#pragma unroll
      for (int ct = 0; ct < 8; ++ct) {
        v16h Bf = load_bfragT(wlsT[3], 32, 0, ct * 16);
        acc[ct] = wmma_f16(A, Bf, acc[ct]);
        float bias = bls[3][ct * 16 + cN];
#pragma unroll
        for (int r = 0; r < 8; ++r) acc[ct][r] += bias;
      }
    }
#pragma unroll
    for (int r = 0; r < 8; ++r) {
      float sum = 0.0f, sq = 0.0f;
#pragma unroll
      for (int ct = 0; ct < 8; ++ct) { float v = acc[ct][r]; sum += v; sq += v * v; }
#pragma unroll
      for (int off = 1; off < 16; off <<= 1) {
        sum += __shfl_xor(sum, off, 32);
        sq += __shfl_xor(sq, off, 32);
      }
      float mu = sum * (1.0f / 128.0f);
      float var = sq * (1.0f / 128.0f) - mu * mu;
      float rstd = rsqrtf(var + 1e-5f);
      int n = qt * 16 + r + rOff;
      _Float16* orow = hout + (((size_t)b * kN + n) * kT + t) * kHID;
#pragma unroll
      for (int ct = 0; ct < 8; ++ct) {
        int cI = ct * 16 + cN;
        float v = (acc[ct][r] - mu) * rstd * lng[cI] + lnb[cI];
        v = v > 0.0f ? v : (__expf(v) - 1.0f);
        orow[cI] = (_Float16)v;
      }
    }
  }
}

// ------------- att1: temporal attention over T per (b,n) + mean-pool -------
__global__ __launch_bounds__(256) void att1_kernel(
    const _Float16* __restrict__ hin, const float* __restrict__ Wq,
    const float* __restrict__ bq, const float* __restrict__ Wk,
    const float* __restrict__ bk, const float* __restrict__ Wv,
    const float* __restrict__ bv, const float* __restrict__ Ws,
    const float* __restrict__ bs, const float* __restrict__ g_,
    const float* __restrict__ b_, float* __restrict__ hpool) {
  __shared__ __attribute__((aligned(16))) _Float16 xs[kT * kHID];
  __shared__ __attribute__((aligned(16))) _Float16 wlsT[4][kHID * kHID]; // [c][k]
  __shared__ float bls[4][kHID];
  __shared__ __attribute__((aligned(16))) _Float16 qs[kT * kHID];
  __shared__ __attribute__((aligned(16))) _Float16 ks[kT * kHID];
  __shared__ __attribute__((aligned(16))) _Float16 vsT[kHID * kT];      // [c][key]
  __shared__ float mcol[kT];
  __shared__ float zcol[kT];
  __shared__ float lng[kHID];
  __shared__ float lnb[kHID];
  __shared__ __attribute__((aligned(16))) _Float16 pst[8][16 * 32];
  __shared__ float pool4[4][kHID];

  const int tid = threadIdx.x;
  const int wave = tid >> 5, lane = tid & 31;
  const int cN = lane & 15;
  const int rOff = (lane >> 4) << 3;
  const int bn = blockIdx.x;

  for (int i = tid; i < kT * kHID / 8; i += 256)  // vectorized b128 copy
    ((v8h*)xs)[i] = ((const v8h*)hin)[(size_t)bn * (kT * kHID / 8) + i];
  {
    const float* Wp[4] = {Wq, Wk, Wv, Ws};
    const float* bp[4] = {bq, bk, bv, bs};
#pragma unroll
    for (int m = 0; m < 4; ++m) {
      for (int i = tid; i < kHID * kHID; i += 256) {  // coalesced read, LDS transpose
        int r = i >> 7, c = i & 127;
        wlsT[m][c * kHID + r] = (_Float16)Wp[m][i];
      }
      if (tid < kHID) bls[m][tid] = bp[m][tid];
    }
    if (tid < kHID) { lng[tid] = g_[tid]; lnb[tid] = b_[tid]; }
  }
  __syncthreads();

  // Q/K/V: 3 mats * 4 mtiles * 8 ctiles = 96 tiles, K=128 (4 chunks)
  for (int tI = wave; tI < 96; tI += 8) {
    int m = tI >> 5, tt = tI & 31;
    int mt = tt >> 3, ct = tt & 7;
    float bias = bls[m][ct * 16 + cN];
    v8f acc;
#pragma unroll
    for (int r = 0; r < 8; ++r) acc[r] = bias;
#pragma unroll
    for (int cc = 0; cc < 4; ++cc) {
      v16h A = load_afrag(xs, kHID, mt * 16, cc * 32);
      v16h Bf = load_bfragT(wlsT[m], kHID, cc * 32, ct * 16);
      acc = wmma_f16(A, Bf, acc);
    }
    if (m < 2) {
      _Float16* dst = (m == 0) ? qs : ks;
#pragma unroll
      for (int r = 0; r < 8; ++r)
        dst[(mt * 16 + r + rOff) * kHID + ct * 16 + cN] = (_Float16)acc[r];
    } else {
      v8h pk;
#pragma unroll
      for (int r = 0; r < 8; ++r) pk[r] = (_Float16)acc[r];
      *(v8h*)(vsT + (ct * 16 + cN) * kT + mt * 16 + rOff) = pk;
    }
  }
  __syncthreads();

  if (wave < 4) {  // per-key stats, 4 key tiles
    int kt = wave;
    float mloc = -3.0e38f, zloc = 0.0f;
    for (int qt = 0; qt < 4; ++qt) {
      v8f s;
#pragma unroll
      for (int r = 0; r < 8; ++r) s[r] = 0.0f;
#pragma unroll
      for (int cc = 0; cc < 4; ++cc) {
        v16h A = load_afrag(qs, kHID, qt * 16, cc * 32);
        v16h Bf = load_bfragT(ks, kHID, cc * 32, kt * 16);
        s = wmma_f16(A, Bf, s);
      }
      float sm = -3.0e38f;
#pragma unroll
      for (int r = 0; r < 8; ++r) { s[r] *= kScl; sm = fmaxf(sm, s[r]); }
      float mn = fmaxf(mloc, sm);
      float zz = zloc * __expf(mloc - mn);
#pragma unroll
      for (int r = 0; r < 8; ++r) zz += __expf(s[r] - mn);
      mloc = mn; zloc = zz;
    }
    float mo = __shfl_xor(mloc, 16, 32);
    float zo = __shfl_xor(zloc, 16, 32);
    float M = fmaxf(mloc, mo);
    float Z = zloc * __expf(mloc - M) + zo * __expf(mo - M);
    if (lane < 16) { mcol[kt * 16 + lane] = M; zcol[kt * 16 + lane] = Z; }
  }
  __syncthreads();

  for (int i = tid; i < kHID * kT; i += 256)
    vsT[i] = (_Float16)((float)vsT[i] / zcol[i & (kT - 1)]);
  __syncthreads();

  if (wave < 4) {  // out rows for q-tile = wave; fuse LN+ELU+temporal mean
    int qt = wave;
    v8f acc[8];
#pragma unroll
    for (int ct = 0; ct < 8; ++ct)
#pragma unroll
      for (int r = 0; r < 8; ++r) acc[ct][r] = 0.0f;
#pragma unroll
    for (int kc = 0; kc < 2; ++kc) {
#pragma unroll
      for (int h = 0; h < 2; ++h) {
        int kt = kc * 2 + h;
        v8f s;
#pragma unroll
        for (int r = 0; r < 8; ++r) s[r] = 0.0f;
#pragma unroll
        for (int cc = 0; cc < 4; ++cc) {
          v16h A = load_afrag(qs, kHID, qt * 16, cc * 32);
          v16h Bf = load_bfragT(ks, kHID, cc * 32, kt * 16);
          s = wmma_f16(A, Bf, s);
        }
        float mc = mcol[kt * 16 + cN];
#pragma unroll
        for (int r = 0; r < 8; ++r)
          pst[wave][(r + rOff) * 32 + h * 16 + cN] =
              (_Float16)__expf(s[r] * kScl - mc);
      }
      v16h pA = load_afrag(pst[wave], 32, 0, 0);
#pragma unroll
      for (int ct = 0; ct < 8; ++ct) {
        v16h Bf = load_bfragT(vsT, kT, kc * 32, ct * 16);
        acc[ct] = wmma_f16(pA, Bf, acc[ct]);
      }
    }
#pragma unroll
    for (int cc = 0; cc < 4; ++cc) {  // skip
      v16h A = load_afrag(xs, kHID, qt * 16, cc * 32);
#pragma unroll
      for (int ct = 0; ct < 8; ++ct) {
        v16h Bf = load_bfragT(wlsT[3], kHID, cc * 32, ct * 16);
        acc[ct] = wmma_f16(A, Bf, acc[ct]);
      }
    }
    float colp[8];
#pragma unroll
    for (int ct = 0; ct < 8; ++ct) {
      float bias = bls[3][ct * 16 + cN];
#pragma unroll
      for (int r = 0; r < 8; ++r) acc[ct][r] += bias;
      colp[ct] = 0.0f;
    }
#pragma unroll
    for (int r = 0; r < 8; ++r) {
      float sum = 0.0f, sq = 0.0f;
#pragma unroll
      for (int ct = 0; ct < 8; ++ct) { float v = acc[ct][r]; sum += v; sq += v * v; }
#pragma unroll
      for (int off = 1; off < 16; off <<= 1) {
        sum += __shfl_xor(sum, off, 32);
        sq += __shfl_xor(sq, off, 32);
      }
      float mu = sum * (1.0f / 128.0f);
      float var = sq * (1.0f / 128.0f) - mu * mu;
      float rstd = rsqrtf(var + 1e-5f);
#pragma unroll
      for (int ct = 0; ct < 8; ++ct) {
        int cI = ct * 16 + cN;
        float v = (acc[ct][r] - mu) * rstd * lng[cI] + lnb[cI];
        v = v > 0.0f ? v : (__expf(v) - 1.0f);
        colp[ct] += v;
      }
    }
#pragma unroll
    for (int ct = 0; ct < 8; ++ct) {
      float vtot = colp[ct] + __shfl_xor(colp[ct], 16, 32);
      if (lane < 16) pool4[wave][ct * 16 + lane] = vtot;
    }
  }
  __syncthreads();
  if (tid < kHID)
    hpool[(size_t)bn * kHID + tid] =
        (pool4[0][tid] + pool4[1][tid] + pool4[2][tid] + pool4[3][tid]) *
        (1.0f / 64.0f);
}

// ----------- head: squad pooling + pairwise product + sigmoid --------------
__global__ __launch_bounds__(256) void head_kernel(
    const float* __restrict__ hpool, const int* __restrict__ e2s,
    const int* __restrict__ s2s, const float* __restrict__ aW,
    const float* __restrict__ ab, float* __restrict__ out) {
  __shared__ float hs[kNSQ][kHID];
  __shared__ float cnt[kNSQ];
  const int tid = threadIdx.x;
  const int b = blockIdx.x;
  if (tid < kNSQ) {
    float c = 0.0f;
    for (int n = 0; n < kN; ++n) c += (e2s[n] == tid) ? 1.0f : 0.0f;
    cnt[tid] = fmaxf(c, 1.0f);
  }
  if (tid < kHID) {
    float acc[kNSQ];
#pragma unroll
    for (int s = 0; s < kNSQ; ++s) acc[s] = 0.0f;
    for (int n = 0; n < kN; ++n)
      acc[e2s[n]] += hpool[((size_t)b * kN + n) * kHID + tid];
#pragma unroll
    for (int s = 0; s < kNSQ; ++s) hs[s][tid] = acc[s];
  }
  __syncthreads();
  if (tid < 64) {
    int i = s2s[tid], j = s2s[64 + tid];
    float logit = ab[0];
    for (int c = 0; c < kHID; ++c)
      logit += (hs[i][c] / cnt[i]) * (hs[j][c] / cnt[j]) * aW[c];
    out[(size_t)b * 64 + tid] = 1.0f / (1.0f + __expf(-logit));
  }
}

extern "C" void kernel_launch(void* const* d_in, const int* in_sizes, int n_in,
                              void* d_out, int out_size, void* d_ws,
                              size_t ws_size, hipStream_t stream) {
  (void)in_sizes; (void)n_in; (void)out_size; (void)ws_size;
  const float* x = (const float*)d_in[0];
  const int* e2s = (const int*)d_in[1];
  const int* s2s = (const int*)d_in[2];
  const float* p[22];
  for (int i = 0; i < 22; ++i) p[i] = (const float*)d_in[3 + i];
  // p[0..9]:  att0 qW,qb,kW,kb,vW,vb,skW,skb,ln_g,ln_b
  // p[10..19]: att1 same order; p[20],p[21]: atk W,b
  _Float16* hws = (_Float16*)d_ws;  // [B,N,T,HID] f16 (att0 out, transposed)
  float* hpool =
      (float*)((char*)d_ws + (size_t)kB * kN * kT * kHID * sizeof(_Float16));
  float* out = (float*)d_out;

  att0_kernel<<<kB * kT, 256, 0, stream>>>(x, p[0], p[1], p[2], p[3], p[4],
                                           p[5], p[6], p[7], p[8], p[9], hws);
  att1_kernel<<<kB * kN, 256, 0, stream>>>(hws, p[10], p[11], p[12], p[13],
                                           p[14], p[15], p[16], p[17], p[18],
                                           p[19], hpool);
  head_kernel<<<kB, 256, 0, stream>>>(hpool, e2s, s2s, p[20], p[21], out);
}
